// ZoneOutLSTM_62792421867990
// MI455X (gfx1250) — matrix operand order repaired and checked by reading, not compiled, not run on hardware
//
#include <hip/hip_runtime.h>
#include <hip/hip_bf16.h>

// ---------------------------------------------------------------------------
// ZoneOut 2-layer LSTM for MI455X (gfx1250).
// T=512, B=32, D=512, H=1024, L=2. KEEP = 0.9.
//
// bf16 weights (L2-resident: 28 MB), fp32 accumulate via
// v_wmma_f32_16x16x32_bf16. Per step, per layer: one concatenated-K GEMM
// ([x_t|h0] K=1536, [h0_pre|h1] K=2048) + one fused cell/zoneout kernel.
// A (32 x K activations) is staged in LDS per workgroup (8x reuse),
// weights stream straight from L2 with compile-time strides.
// ---------------------------------------------------------------------------

#define T_STEPS 512
#define BATCH   32
#define DIM     512
#define HID     1024
#define FOURH   4096
#define KEEPP   0.9f
#define KC      256          // K-chunk staged in LDS (32 x 256 bf16 = 16 KB)
#define APAD    8            // LDS row padding (bf16 elems) to spread banks

typedef __attribute__((ext_vector_type(16))) __bf16 v16bf;
typedef __attribute__((ext_vector_type(8)))  float  v8f;

__device__ __forceinline__ unsigned short f2bf(float f) {
    unsigned u = __float_as_uint(f);
    unsigned r = u + 0x7FFFu + ((u >> 16) & 1u);   // round-to-nearest-even
    return (unsigned short)(r >> 16);
}

__device__ __forceinline__ v8f bfwmma(const v16bf& a, const v16bf& b, const v8f& c) {
    return __builtin_amdgcn_wmma_f32_16x16x32_bf16(false, a, false, b,
                                                   (short)0, c, false, false);
}

// ---------------------------------------------------------------------------
// One K-segment of the GEMM: stage A chunk-by-chunk in LDS, stream W rows
// from global with immediate offsets (all strides compile-time).
// ---------------------------------------------------------------------------
template<int KSEG, int LDA, int LDW>
__device__ __forceinline__ void gemm_seg(const unsigned short* __restrict__ Aseg,
                                         const unsigned short* __restrict__ wrow0,
                                         const unsigned short* __restrict__ wrow1,
                                         int wk0,
                                         unsigned short (&As)[32][KC + APAD],
                                         v8f& acc00, v8f& acc01,
                                         v8f& acc10, v8f& acc11)
{
    const int tid  = threadIdx.x;
    const int lane = tid & 31;
    const int ln   = lane & 15;
    const int kh   = lane >> 4;

    for (int kc = 0; kc < KSEG; kc += KC) {
        __syncthreads();
        // cooperative stage: 32*KC bf16 = 1024 uint4, 256 threads x 4
#pragma unroll
        for (int i = 0; i < (32 * KC / 8) / 256; ++i) {
            int u4  = tid + i * 256;
            int row = u4 >> 5;                 // KC/8 = 32 uint4 per row
            int col = (u4 & 31) * 8;
            *(uint4*)&As[row][col] =
                *(const uint4*)(Aseg + (size_t)row * LDA + kc + col);
        }
        __syncthreads();

#pragma unroll
        for (int kb = 0; kb < KC; kb += 32) {
            union { v16bf v; uint4 u[2]; } a0, a1, b0, b1;
            // 16-bit A frag: VGPR0..3 = K[8*kh..+7], VGPR4..7 = K[16+8*kh..+7]
            a0.u[0] = *(const uint4*)&As[ln][kb + 8 * kh];
            a0.u[1] = *(const uint4*)&As[ln][kb + 16 + 8 * kh];
            a1.u[0] = *(const uint4*)&As[16 + ln][kb + 8 * kh];
            a1.u[1] = *(const uint4*)&As[16 + ln][kb + 16 + 8 * kh];
            // 32x16 B frag: VGPR0..3 = K[8*kh..+7], VGPR4..7 = K[16+8*kh..+7] of weight row
            const int wo = wk0 + kc + kb + 8 * kh;
            b0.u[0] = *(const uint4*)(wrow0 + wo);
            b0.u[1] = *(const uint4*)(wrow0 + wo + 16);
            b1.u[0] = *(const uint4*)(wrow1 + wo);
            b1.u[1] = *(const uint4*)(wrow1 + wo + 16);

            acc00 = bfwmma(a0.v, b0.v, acc00);
            acc01 = bfwmma(a0.v, b1.v, acc01);
            acc10 = bfwmma(a1.v, b0.v, acc10);
            acc11 = bfwmma(a1.v, b1.v, acc11);
        }
    }
}

// ---------------------------------------------------------------------------
// gates[32,4096] = [A0 | A1] @ W^T ; W is [4096, K0+K1] row-major bf16.
// Wave tile 32x32 (4 WMMAs / 32-K step); 8 waves/block, 16 blocks -> N=4096.
// ---------------------------------------------------------------------------
template<int K0, int LD0, int K1, int LD1, int LDW>
__global__ __launch_bounds__(256)
void gemm_gates(const unsigned short* __restrict__ A0,
                const unsigned short* __restrict__ A1,
                const unsigned short* __restrict__ W,
                float* __restrict__ gates)
{
    __shared__ unsigned short As[32][KC + APAD];

    const int lane  = threadIdx.x & 31;
    const int wave  = threadIdx.x >> 5;
    const int nbase = (blockIdx.x * 8 + wave) * 32;
    const int ln    = lane & 15;
    const int kh    = lane >> 4;

    v8f acc00 = {}, acc01 = {}, acc10 = {}, acc11 = {};
    const unsigned short* wrow0 = W + (size_t)(nbase + ln) * LDW;
    const unsigned short* wrow1 = W + (size_t)(nbase + 16 + ln) * LDW;

    gemm_seg<K0, LD0, LDW>(A0, wrow0, wrow1, 0, As, acc00, acc01, acc10, acc11);
    if constexpr (K1 > 0)
        gemm_seg<K1, LD1, LDW>(A1, wrow0, wrow1, K0, As, acc00, acc01, acc10, acc11);

    // C layout: VGPR r -> (M = r + 8*kh, N = ln)
    float* g0 = gates + nbase + ln;
    float* g1 = gates + nbase + 16 + ln;
#pragma unroll
    for (int r = 0; r < 8; ++r) {
        const size_t m0 = (size_t)(r + 8 * kh) * FOURH;
        const size_t m1 = (size_t)(16 + r + 8 * kh) * FOURH;
        g0[m0] = acc00[r];
        g1[m0] = acc01[r];
        g0[m1] = acc10[r];
        g1[m1] = acc11[r];
    }
}

// ---------------------------------------------------------------------------
// Fused LSTM cell + zoneout. One thread per (b,h) of 32x1024.
// ---------------------------------------------------------------------------
__global__ __launch_bounds__(256)
void lstm_cell(const float* __restrict__ gates, const float* __restrict__ bias,
               float* __restrict__ h_state, float* __restrict__ c_state,
               const float* __restrict__ uh, const float* __restrict__ uc,
               unsigned short* __restrict__ hz_bf, int hz_ld,
               unsigned short* __restrict__ hpre_bf, int hpre_ld,
               float* __restrict__ out_pre)
{
    const int idx = blockIdx.x * blockDim.x + threadIdx.x;   // 0..32767
    const int b  = idx >> 10;
    const int hh = idx & 1023;

    const float* g = gates + (size_t)b * FOURH + hh;
    float gi = g[0]       + bias[hh];
    float gf = g[HID]     + bias[HID + hh];
    float gg = g[2 * HID] + bias[2 * HID + hh];
    float go = g[3 * HID] + bias[3 * HID + hh];

    float i = 1.0f / (1.0f + expf(-gi));
    float f = 1.0f / (1.0f + expf(-gf));
    float gt = tanhf(gg);
    float o = 1.0f / (1.0f + expf(-go));

    float c_old = c_state[idx];
    float c_new = f * c_old + i * gt;
    float h_new = o * tanhf(c_new);
    float h_old = h_state[idx];

    float hn = (uh[idx] < KEEPP) ? h_new : h_old;   // zoneout select
    float cn = (uc[idx] < KEEPP) ? c_new : c_old;

    h_state[idx] = hn;
    c_state[idx] = cn;
    hz_bf[(size_t)b * hz_ld + hh] = f2bf(hn);
    if (hpre_bf) hpre_bf[(size_t)b * hpre_ld + hh] = f2bf(h_new);
    if (out_pre) out_pre[idx] = h_new;
}

// --------------------------- prep kernels ----------------------------------
__global__ void pack_cat_bf16(const float* __restrict__ Wa, int Ka,
                              const float* __restrict__ Wb, int Kb,
                              unsigned short* __restrict__ out, int total)
{
    int idx = blockIdx.x * blockDim.x + threadIdx.x;
    if (idx >= total) return;
    const int K = Ka + Kb;
    const int n = idx / K;
    const int k = idx - n * K;
    float v = (k < Ka) ? Wa[(size_t)n * Ka + k] : Wb[(size_t)n * Kb + (k - Ka)];
    out[idx] = f2bf(v);
}

__global__ void bias_sum(const float* __restrict__ a0, const float* __restrict__ a1,
                         const float* __restrict__ b0, const float* __restrict__ b1,
                         float* __restrict__ o0, float* __restrict__ o1)
{
    int idx = blockIdx.x * blockDim.x + threadIdx.x;
    if (idx >= FOURH) return;
    o0[idx] = a0[idx] + a1[idx];
    o1[idx] = b0[idx] + b1[idx];
}

__global__ void cvt_f32_bf16(const float* __restrict__ in,
                             unsigned short* __restrict__ out, int n)
{
    int idx = blockIdx.x * blockDim.x + threadIdx.x;
    if (idx < n) out[idx] = f2bf(in[idx]);
}

__global__ void final_copy(const float* __restrict__ h0, const float* __restrict__ h1,
                           const float* __restrict__ c0, const float* __restrict__ c1,
                           float* __restrict__ dst)
{
    int idx = blockIdx.x * blockDim.x + threadIdx.x;   // 0..131071
    int seg = idx >> 15, off = idx & 32767;
    const float* src = (seg == 0) ? h0 : (seg == 1) ? h1 : (seg == 2) ? c0 : c1;
    dst[idx] = src[off];
}

// ---------------------------------------------------------------------------
extern "C" void kernel_launch(void* const* d_in, const int* in_sizes, int n_in,
                              void* d_out, int out_size, void* d_ws, size_t ws_size,
                              hipStream_t stream)
{
    const float* x    = (const float*)d_in[0];
    const float* Wih0 = (const float*)d_in[1];
    const float* Whh0 = (const float*)d_in[2];
    const float* bih0 = (const float*)d_in[3];
    const float* bhh0 = (const float*)d_in[4];
    const float* Wih1 = (const float*)d_in[5];
    const float* Whh1 = (const float*)d_in[6];
    const float* bih1 = (const float*)d_in[7];
    const float* bhh1 = (const float*)d_in[8];
    const float* u_h  = (const float*)d_in[9];
    const float* u_c  = (const float*)d_in[10];
    float* out = (float*)d_out;

    // ---- workspace layout (bytes) ----
    char* ws = (char*)d_ws;
    size_t off = 0;
    unsigned short* W0cat = (unsigned short*)(ws + off); off += (size_t)FOURH * 1536 * 2; // 12 MB
    unsigned short* W1cat = (unsigned short*)(ws + off); off += (size_t)FOURH * 2048 * 2; // 16 MB
    unsigned short* xbf   = (unsigned short*)(ws + off); off += (size_t)T_STEPS * BATCH * DIM * 2; // 16 MB
    float* bias0  = (float*)(ws + off); off += FOURH * 4;
    float* bias1  = (float*)(ws + off); off += FOURH * 4;
    float* gates0 = (float*)(ws + off); off += (size_t)BATCH * FOURH * 4;
    float* gates1 = (float*)(ws + off); off += (size_t)BATCH * FOURH * 4;
    // ---- zero-initialized region (states + recurrent bf16 A slots) ----
    char* zbase = ws + off;
    float* h0st = (float*)(ws + off); off += (size_t)BATCH * HID * 4;
    float* c0st = (float*)(ws + off); off += (size_t)BATCH * HID * 4;
    float* h1st = (float*)(ws + off); off += (size_t)BATCH * HID * 4;
    float* c1st = (float*)(ws + off); off += (size_t)BATCH * HID * 4;
    unsigned short* h0z = (unsigned short*)(ws + off); off += (size_t)BATCH * HID * 2;       // [32,1024]
    unsigned short* A1  = (unsigned short*)(ws + off); off += (size_t)BATCH * 2 * HID * 2;   // [32,2048]
    size_t zbytes = (size_t)(ws + off - zbase);

    // ---- one-time prep (re-done every call: deterministic, capture-safe) ----
    hipMemsetAsync(zbase, 0, zbytes, stream);
    {
        int tot0 = FOURH * 1536;
        pack_cat_bf16<<<(tot0 + 255) / 256, 256, 0, stream>>>(Wih0, DIM, Whh0, HID, W0cat, tot0);
        int tot1 = FOURH * 2048;
        pack_cat_bf16<<<(tot1 + 255) / 256, 256, 0, stream>>>(Wih1, HID, Whh1, HID, W1cat, tot1);
        bias_sum<<<(FOURH + 255) / 256, 256, 0, stream>>>(bih0, bhh0, bih1, bhh1, bias0, bias1);
        int nx = T_STEPS * BATCH * DIM;
        cvt_f32_bf16<<<(nx + 255) / 256, 256, 0, stream>>>(x, xbf, nx);
    }

    const int BH = BATCH * HID;                 // 32768
    // ---- sequential recurrence ----
    for (int t = 0; t < T_STEPS; ++t) {
        const unsigned short* xt = xbf + (size_t)t * BATCH * DIM;
        // layer 0: gates0 = [x_t | h0z] @ [Wih0|Whh0]^T   (K = 512 + 1024)
        gemm_gates<DIM, DIM, HID, HID, 1536><<<16, 256, 0, stream>>>(xt, h0z, W0cat, gates0);
        // cell 0: h0z <- post-zoneout bf16 ; A1[:, :1024] <- pre-zoneout bf16
        lstm_cell<<<BH / 256, 256, 0, stream>>>(gates0, bias0, h0st, c0st,
                                                u_h + (size_t)(t * 2 + 0) * BH,
                                                u_c + (size_t)(t * 2 + 0) * BH,
                                                h0z, HID, A1, 2 * HID, nullptr);
        // layer 1: gates1 = [h0_pre | h1z] @ [Wih1|Whh1]^T  (K = 2048, one buffer)
        gemm_gates<2 * HID, 2 * HID, 0, 1, 2048><<<16, 256, 0, stream>>>(A1, A1, W1cat, gates1);
        // cell 1: A1[:, 1024:] <- post-zoneout bf16 ; outputs[t] <- pre-zoneout fp32
        lstm_cell<<<BH / 256, 256, 0, stream>>>(gates1, bias1, h1st, c1st,
                                                u_h + (size_t)(t * 2 + 1) * BH,
                                                u_c + (size_t)(t * 2 + 1) * BH,
                                                A1 + HID, 2 * HID, nullptr, 0,
                                                out + (size_t)t * BH);
    }

    // ---- h_n [2,B,H] then c_n [2,B,H] after outputs ----
    final_copy<<<(4 * BH) / 256, 256, 0, stream>>>(h0st, h1st, c0st, c1st,
                                                   out + (size_t)T_STEPS * BH);
}
